// MultiHeadSelfAttention_66374424592676
// MI455X (gfx1250) — matrix operand order repaired and checked by reading, not compiled, not run on hardware
//
#include <hip/hip_runtime.h>
#include <cstdint>

// ---------------------------------------------------------------------------
// MHSA for MI455X (gfx1250): bf16 WMMA GEMMs + flash attention.
// B=1, N=4096, D=768, 12 heads, head_dim=64.
// ---------------------------------------------------------------------------

typedef __bf16 bf16;
typedef __attribute__((ext_vector_type(16))) __bf16 bf16x16;
typedef __attribute__((ext_vector_type(8)))  float   f32x8;

union F16x16 {
    bf16x16 v;
    uint4   u[2];
};

#define TOK    4096
#define DMODEL 768
#define NHEAD  12
#define HD     64

__device__ __forceinline__ f32x8 wmma_bf16(bf16x16 a, bf16x16 b, f32x8 c) {
    // D = A(16x32 bf16) * B(32x16 bf16) + C(16x16 f32)
    return __builtin_amdgcn_wmma_f32_16x16x32_bf16(false, a, false, b,
                                                   (short)0, c, false, false);
}

// DPP16 ROW_XMASK lane-xor within 16-lane rows: pure-VALU butterfly step
// (replaces ds_bpermute + s_wait_dscnt chains for the softmax row max).
template <int XMASK>
__device__ __forceinline__ float xor16_f32(float x) {
    return __int_as_float(__builtin_amdgcn_update_dpp(
        0, __float_as_int(x), 0x160 | XMASK, 0xF, 0xF, true));
}

// ---------------------------------------------------------------------------
// CDNA5 async global->LDS copies (ASYNCcnt-tracked, no VGPR round trip).
// Builtin takes v4i32* pointers: global src in AS1, LDS dst in AS3.
// Guarded: falls back to sync loads if the toolchain lacks the builtins.
// ---------------------------------------------------------------------------
#if defined(__HIP_DEVICE_COMPILE__) && \
    __has_builtin(__builtin_amdgcn_global_load_async_to_lds_b128) && \
    __has_builtin(__builtin_amdgcn_s_wait_asynccnt)
#define GEMM_ASYNC 1
#else
#define GEMM_ASYNC 0
#endif

#if GEMM_ASYNC
typedef __attribute__((__vector_size__(4 * sizeof(int)))) int v4i;
typedef __attribute__((address_space(1))) v4i as1_v4i;
typedef __attribute__((address_space(3))) v4i as3_v4i;
__device__ __forceinline__ void async_cp16(const bf16* g, bf16* l) {
    // generic LDS pointer: low 32 bits are the LDS byte offset (aperture rule)
    __builtin_amdgcn_global_load_async_to_lds_b128(
        (as1_v4i*)(uintptr_t)g,
        (as3_v4i*)(uint32_t)(uintptr_t)l,
        0, 0);
}
#endif

// ---------------------------------------------------------------------------
// fp32 -> bf16 elementwise convert
// ---------------------------------------------------------------------------
__global__ void cvt_bf16_kernel(const float* __restrict__ in,
                                bf16* __restrict__ out, int n) {
    int i = blockIdx.x * blockDim.x + threadIdx.x;
    if (i < n) out[i] = (bf16)in[i];
}

// ---------------------------------------------------------------------------
// W[in][out] fp32 -> Wt[out][in] bf16 (tiled transpose via LDS)
// ---------------------------------------------------------------------------
__global__ void transpose_cvt_kernel(const float* __restrict__ W,
                                     bf16* __restrict__ Wt) {
    __shared__ float tile[32][33];
    const int bx = blockIdx.x * 32;   // out-dim tile
    const int by = blockIdx.y * 32;   // in-dim tile
    const int tx = threadIdx.x;       // 0..31
    const int ty = threadIdx.y;       // 0..7
    #pragma unroll
    for (int r = 0; r < 32; r += 8)
        tile[ty + r][tx] = W[(size_t)(by + ty + r) * DMODEL + bx + tx];
    __syncthreads();
    #pragma unroll
    for (int r = 0; r < 32; r += 8)
        Wt[(size_t)(bx + ty + r) * DMODEL + by + tx] = (bf16)tile[tx][ty + r];
}

// ---------------------------------------------------------------------------
// C[m][n] = (sum_k A[m][k] * Bt[n][k] + bias[n]) * scale
// A: [4096][768] bf16 row-major, Bt: [768][768] bf16 (pre-transposed weight).
// MODE 0: out bf16  [h][4096][64]   (Q / K layout, head-major)
// MODE 1: out bf16  [h][64][4096]   (V transposed; packed b128 stores)
// MODE 2: out fp32  [4096][768]     (final output)
// Block: 256 threads = 8 waves; tile BM=128 x BN=64, BK=64.
// Async path: double-buffered GLOBAL_LOAD_ASYNC_TO_LDS_B128 staging,
// s_wait_asynccnt 6 overlaps next-tile DMA with current-tile WMMA.
// ---------------------------------------------------------------------------
template <int MODE>
__global__ __launch_bounds__(256)
void gemm_kernel(const bf16* __restrict__ A, const bf16* __restrict__ Bt,
                 const float* __restrict__ bias, void* __restrict__ outp,
                 float scale)
{
    constexpr int BM = 128, BN = 64, BK = 64;
    constexpr int LD = BK + 8;               // +16B row pad: conflict-free frag reads
#if GEMM_ASYNC
    constexpr int NB = 2;
#else
    constexpr int NB = 1;
#endif
    __shared__ __align__(16) bf16 As[NB][BM][LD];
    __shared__ __align__(16) bf16 Bs[NB][BN][LD];

    const int tid  = threadIdx.x;
    const int lane = tid & 31;
    const int wave = tid >> 5;
    const int wm   = wave >> 1;              // 0..3  (M sub-tile)
    const int wn   = wave & 1;               // 0..1  (N sub-tile)
    const int l15  = lane & 15;
    const int hs   = lane >> 4;

    const int mblk = blockIdx.x * BM;
    const int nblk = blockIdx.y * BN;

    // cooperative-load assignments
    const int ar   = tid & (BM - 1);         // A row
    const int ah   = tid >> 7;               // which 32-half chunk of the row
    const int br   = tid >> 2;               // B row (0..63)
    const int bseg = tid & 3;                // which 16-half chunk

    const bf16* gAbase = A  + (size_t)(mblk + ar) * DMODEL + ah * 32;
    const bf16* gBbase = Bt + (size_t)(nblk + br) * DMODEL + bseg * 16;

    f32x8 acc[2][2] = {};

#if GEMM_ASYNC
    // 6 async b128 issues per thread per tile (4 for A, 2 for B)
    {
        #pragma unroll
        for (int c = 0; c < 4; ++c)
            async_cp16(gAbase + 0 + c * 8, &As[0][ar][ah * 32 + c * 8]);
        #pragma unroll
        for (int c = 0; c < 2; ++c)
            async_cp16(gBbase + 0 + c * 8, &Bs[0][br][bseg * 16 + c * 8]);
    }
#endif

    for (int kb = 0; kb < DMODEL; kb += BK) {
#if GEMM_ASYNC
        const int buf = (kb / BK) & 1;
        if (kb + BK < DMODEL) {
            const int nbuf = buf ^ 1;
            #pragma unroll
            for (int c = 0; c < 4; ++c)
                async_cp16(gAbase + kb + BK + c * 8, &As[nbuf][ar][ah * 32 + c * 8]);
            #pragma unroll
            for (int c = 0; c < 2; ++c)
                async_cp16(gBbase + kb + BK + c * 8, &Bs[nbuf][br][bseg * 16 + c * 8]);
            __builtin_amdgcn_s_wait_asynccnt(6);   // current tile resident
        } else {
            __builtin_amdgcn_s_wait_asynccnt(0);
        }
#else
        const int buf = 0;
        const uint4* gA = reinterpret_cast<const uint4*>(gAbase + kb);
        uint4* sA = reinterpret_cast<uint4*>(&As[0][ar][ah * 32]);
        sA[0] = gA[0]; sA[1] = gA[1]; sA[2] = gA[2]; sA[3] = gA[3];
        const uint4* gB = reinterpret_cast<const uint4*>(gBbase + kb);
        uint4* sB = reinterpret_cast<uint4*>(&Bs[0][br][bseg * 16]);
        sB[0] = gB[0]; sB[1] = gB[1];
#endif
        __syncthreads();

        #pragma unroll
        for (int kk = 0; kk < BK; kk += 32) {
            F16x16 afrag[2], bfrag[2];
            #pragma unroll
            for (int i = 0; i < 2; ++i) {
                // A-frag (16x32 bf16): lane row = l15; K = 16*hs + e (contiguous 32B)
                const bf16* p = &As[buf][wm * 32 + i * 16 + l15][kk];
                afrag[i].u[0] = *reinterpret_cast<const uint4*>(p + 16 * hs);
                afrag[i].u[1] = *reinterpret_cast<const uint4*>(p + 16 * hs + 8);
            }
            #pragma unroll
            for (int j = 0; j < 2; ++j) {
                // B-frag (32x16 bf16): lane col = l15; K = 16*hs + e (contiguous 32B)
                const bf16* p = &Bs[buf][wn * 32 + j * 16 + l15][kk + 16 * hs];
                bfrag[j].u[0] = *reinterpret_cast<const uint4*>(p);
                bfrag[j].u[1] = *reinterpret_cast<const uint4*>(p + 8);
            }
            #pragma unroll
            for (int i = 0; i < 2; ++i)
                #pragma unroll
                for (int j = 0; j < 2; ++j)
                    acc[i][j] = wmma_bf16(afrag[i].v, bfrag[j].v, acc[i][j]);
        }
        __syncthreads();
    }

    // epilogue: C-frag layout is M = 8*hs + v, N = l15
    #pragma unroll
    for (int j = 0; j < 2; ++j) {
        const int col = nblk + wn * 32 + j * 16 + l15;
        const float bcol = bias[col];
        #pragma unroll
        for (int i = 0; i < 2; ++i) {
            const int r0 = mblk + wm * 32 + i * 16 + 8 * hs;
            if constexpr (MODE == 1) {
                // rows r0..r0+7 are contiguous in V^T layout -> one b128 store
                union { bf16 h[8]; uint4 u; } pk;
                #pragma unroll
                for (int v = 0; v < 8; ++v)
                    pk.h[v] = (bf16)((acc[i][j][v] + bcol) * scale);
                bf16* o = (bf16*)outp;
                const int h = col >> 6, dd = col & 63;
                *reinterpret_cast<uint4*>(&o[((size_t)h * HD + dd) * TOK + r0]) = pk.u;
            } else {
                #pragma unroll
                for (int v = 0; v < 8; ++v) {
                    const int row = r0 + v;
                    const float val = (acc[i][j][v] + bcol) * scale;
                    if constexpr (MODE == 0) {
                        bf16* o = (bf16*)outp;
                        const int h = col >> 6, dd = col & 63;
                        o[((size_t)h * TOK + row) * HD + dd] = (bf16)val;
                    } else {
                        ((float*)outp)[(size_t)row * DMODEL + col] = val;
                    }
                }
            }
        }
    }
}

// ---------------------------------------------------------------------------
// Flash attention, one (head, 64-query-row) block; 4 waves x 16 rows each.
// Q pre-scaled by 0.125*log2(e) so softmax runs on native exp2.
// Per 32-key step: 4 WMMA for S = Q*K^T, online max via DPP ROW_XMASK
// butterflies (pure VALU, no LDS), P through padded LDS (C-layout ->
// A-layout), 4 WMMA for O += P*V, and one extra WMMA against an all-ones
// B-fragment that accumulates the softmax denominator l = P*1 in the
// matrix pipe (no cross-lane sum reductions).
// ---------------------------------------------------------------------------
__global__ __launch_bounds__(128)
void flash_kernel(const bf16* __restrict__ Q, const bf16* __restrict__ K,
                  const bf16* __restrict__ Vt, bf16* __restrict__ Oa)
{
    __shared__ __align__(16) bf16 Plds[4][16][40];   // per-wave 16x32 (+8 pad)
    const int tid  = threadIdx.x;
    const int lane = tid & 31;
    const int wave = tid >> 5;
    const int l15  = lane & 15;
    const int hs   = lane >> 4;
    const int head = blockIdx.y;
    const int q0   = blockIdx.x * 64 + wave * 16;

    const bf16* Qh = Q  + (size_t)head * TOK * HD;
    const bf16* Kh = K  + (size_t)head * TOK * HD;
    const bf16* Vh = Vt + (size_t)head * HD * TOK;

    // Q A-fragments (reused across the whole key loop)
    F16x16 aq[2];
    #pragma unroll
    for (int k2 = 0; k2 < 2; ++k2) {
        const bf16* p = Qh + (size_t)(q0 + l15) * HD + k2 * 32;
        aq[k2].u[0] = *reinterpret_cast<const uint4*>(p + 16 * hs);
        aq[k2].u[1] = *reinterpret_cast<const uint4*>(p + 16 * hs + 8);
    }

    // all-ones B-fragment for the l = P*1 denominator WMMA
    bf16x16 ones;
    #pragma unroll
    for (int e = 0; e < 16; ++e) ones[e] = (bf16)1.0f;

    f32x8 Oacc[4] = {};
    f32x8 Lacc = {};
    float m[8];
    #pragma unroll
    for (int v = 0; v < 8; ++v) m[v] = -3.0e38f;

    for (int kb = 0; kb < TOK; kb += 32) {
        // K^T B-fragments: lane key = kb + 16j + l15; dims contiguous
        F16x16 bk2[2][2];
        #pragma unroll
        for (int j = 0; j < 2; ++j)
            #pragma unroll
            for (int k2 = 0; k2 < 2; ++k2) {
                const bf16* p = Kh + (size_t)(kb + j * 16 + l15) * HD
                                   + k2 * 32 + 16 * hs;
                bk2[j][k2].u[0] = *reinterpret_cast<const uint4*>(p);
                bk2[j][k2].u[1] = *reinterpret_cast<const uint4*>(p + 8);
            }
        f32x8 S0 = {}, S1 = {};
        S0 = wmma_bf16(aq[0].v, bk2[0][0].v, S0);
        S0 = wmma_bf16(aq[1].v, bk2[0][1].v, S0);
        S1 = wmma_bf16(aq[0].v, bk2[1][0].v, S1);
        S1 = wmma_bf16(aq[1].v, bk2[1][1].v, S1);

        // online softmax max (base-2; log2e folded into Q); VALU-only butterfly
        float P0[8], P1[8], alpha[8];
        #pragma unroll
        for (int v = 0; v < 8; ++v) {
            float mx = fmaxf(S0[v], S1[v]);
            mx = fmaxf(mx, xor16_f32<1>(mx));
            mx = fmaxf(mx, xor16_f32<2>(mx));
            mx = fmaxf(mx, xor16_f32<4>(mx));
            mx = fmaxf(mx, xor16_f32<8>(mx));
            const float mn = fmaxf(m[v], mx);
            alpha[v] = exp2f(m[v] - mn);
            m[v] = mn;
            P0[v] = exp2f(S0[v] - mn);
            P1[v] = exp2f(S1[v] - mn);
        }

        // reshape P (C-frag layout) -> A-frag layout through LDS
        #pragma unroll
        for (int v = 0; v < 8; ++v) {
            const int r = 8 * hs + v;
            Plds[wave][r][l15]      = (bf16)P0[v];
            Plds[wave][r][16 + l15] = (bf16)P1[v];
        }
        __syncthreads();

        F16x16 ap;
        ap.u[0] = *reinterpret_cast<const uint4*>(&Plds[wave][l15][16 * hs]);
        ap.u[1] = *reinterpret_cast<const uint4*>(&Plds[wave][l15][16 * hs + 8]);

        // V B-fragments from transposed V: lane d-col = nt*16 + l15, keys contiguous
        F16x16 bv4[4];
        #pragma unroll
        for (int nt = 0; nt < 4; ++nt) {
            const bf16* p = Vh + (size_t)(nt * 16 + l15) * TOK + kb + 16 * hs;
            bv4[nt].u[0] = *reinterpret_cast<const uint4*>(p);
            bv4[nt].u[1] = *reinterpret_cast<const uint4*>(p + 8);
        }
        #pragma unroll
        for (int nt = 0; nt < 4; ++nt) {
            #pragma unroll
            for (int v = 0; v < 8; ++v) Oacc[nt][v] *= alpha[v];
            Oacc[nt] = wmma_bf16(ap.v, bv4[nt].v, Oacc[nt]);
        }
        // denominator: Lacc = Lacc*alpha + rowsum(P) via ones-column WMMA
        #pragma unroll
        for (int v = 0; v < 8; ++v) Lacc[v] *= alpha[v];
        Lacc = wmma_bf16(ap.v, ones, Lacc);

        __syncthreads();
    }

    float inv[8];
    #pragma unroll
    for (int v = 0; v < 8; ++v) inv[v] = 1.0f / Lacc[v];
    #pragma unroll
    for (int nt = 0; nt < 4; ++nt)
        #pragma unroll
        for (int v = 0; v < 8; ++v) {
            const int row = q0 + 8 * hs + v;
            const int col = head * HD + nt * 16 + l15;
            Oa[(size_t)row * DMODEL + col] = (bf16)(Oacc[nt][v] * inv[v]);
        }
}

// ---------------------------------------------------------------------------
// host launcher
// ---------------------------------------------------------------------------
extern "C" void kernel_launch(void* const* d_in, const int* in_sizes, int n_in,
                              void* d_out, int out_size, void* d_ws, size_t ws_size,
                              hipStream_t stream)
{
    (void)in_sizes; (void)n_in; (void)out_size; (void)ws_size;
    const float* x  = (const float*)d_in[0];
    const float* Wq = (const float*)d_in[1];
    const float* bq = (const float*)d_in[2];
    const float* Wk = (const float*)d_in[3];
    const float* bk = (const float*)d_in[4];
    const float* Wv = (const float*)d_in[5];
    const float* bv = (const float*)d_in[6];
    const float* Wo = (const float*)d_in[7];
    const float* bo = (const float*)d_in[8];
    float* out = (float*)d_out;

    char* ws = (char*)d_ws;
    size_t off = 0;
    auto take = [&](size_t bytes) -> void* {
        void* p = ws + off;
        off += (bytes + 255) & ~(size_t)255;
        return p;
    };
    bf16* xb  = (bf16*)take((size_t)TOK * DMODEL * 2);      // x in bf16
    bf16* wqt = (bf16*)take((size_t)DMODEL * DMODEL * 2);   // Wq^T bf16
    bf16* wkt = (bf16*)take((size_t)DMODEL * DMODEL * 2);
    bf16* wvt = (bf16*)take((size_t)DMODEL * DMODEL * 2);
    bf16* wot = (bf16*)take((size_t)DMODEL * DMODEL * 2);
    bf16* Qs  = (bf16*)take((size_t)TOK * DMODEL * 2);      // [h][N][64], pre-scaled
    bf16* Ks  = (bf16*)take((size_t)TOK * DMODEL * 2);      // [h][N][64]
    bf16* Vts = (bf16*)take((size_t)TOK * DMODEL * 2);      // [h][64][N]
    bf16* At  = (bf16*)take((size_t)TOK * DMODEL * 2);      // attn out [N][768]

    const int nx = TOK * DMODEL;
    cvt_bf16_kernel<<<(nx + 255) / 256, 256, 0, stream>>>(x, xb, nx);

    dim3 tb(32, 8), tg(DMODEL / 32, DMODEL / 32);
    transpose_cvt_kernel<<<tg, tb, 0, stream>>>(Wq, wqt);
    transpose_cvt_kernel<<<tg, tb, 0, stream>>>(Wk, wkt);
    transpose_cvt_kernel<<<tg, tb, 0, stream>>>(Wv, wvt);
    transpose_cvt_kernel<<<tg, tb, 0, stream>>>(Wo, wot);

    dim3 gg(TOK / 128, DMODEL / 64), gb(256);
    const float qscale = 0.125f * 1.4426950408889634f;  // 1/sqrt(64) * log2(e)
    gemm_kernel<0><<<gg, gb, 0, stream>>>(xb, wqt, bq, Qs, qscale);
    gemm_kernel<0><<<gg, gb, 0, stream>>>(xb, wkt, bk, Ks, 1.0f);
    gemm_kernel<1><<<gg, gb, 0, stream>>>(xb, wvt, bv, Vts, 1.0f);

    flash_kernel<<<dim3(TOK / 64, NHEAD), 128, 0, stream>>>(Qs, Ks, Vts, At);

    gemm_kernel<2><<<gg, gb, 0, stream>>>(At, wot, bo, (void*)out, 1.0f);
}
